// multi_head_attention_block_12661563588839
// MI455X (gfx1250) — compile-verified, matrix operand order checked by reading, not run on hardware
//
#include <hip/hip_runtime.h>
#include <hip/hip_bf16.h>

typedef _Float16 v8h  __attribute__((ext_vector_type(8)));
typedef _Float16 v16h __attribute__((ext_vector_type(16)));
typedef float    v8f  __attribute__((ext_vector_type(8)));

#define WMMA_F16(A, B, C) \
  __builtin_amdgcn_wmma_f32_16x16x32_f16(false, (A), false, (B), (short)0, (C), false, false)

static __device__ inline v8f zero8f() {
  v8f r;
#pragma unroll
  for (int i = 0; i < 8; ++i) r[i] = 0.0f;
  return r;
}

// Build a 16-half A/B fragment from two contiguous 8-half chunks.
// 16-bit fragment layout (CDNA5): VGPR0..3 = K base chunk (8 contiguous halfs),
// VGPR4..7 = K base+16 chunk; lane>>4 selects the +8 K offset.
static __device__ inline v16h ld_frag(const _Float16* __restrict__ base, int lo_off, int hi_off) {
  v8h lo = *(const v8h*)(base + lo_off);
  v8h hi = *(const v8h*)(base + hi_off);
  v16h r;
#pragma unroll
  for (int i = 0; i < 8; ++i) { r[i] = lo[i]; r[i + 8] = hi[i]; }
  return r;
}

// ---------------------------------------------------------------------------
// Kernel 1: x = LayerNorm(token_embedding + PE); emit f16
// one block per row (b*n = 4096 rows), 256 threads, 4 cols/thread
// ---------------------------------------------------------------------------
__global__ void pe_ln_kernel(const float* __restrict__ emb,
                             const float* __restrict__ gamma,
                             const float* __restrict__ beta,
                             _Float16* __restrict__ xh) {
  const int row = blockIdx.x;          // 0..4095
  const int pos = row & 1023;          // token position
  const int t = threadIdx.x;           // 0..255
  const float inv_d = 1.0f / 1024.0f;

  float vals[4];
  float s = 0.0f, s2 = 0.0f;
#pragma unroll
  for (int j = 0; j < 4; ++j) {
    int c = t * 4 + j;
    int i2 = c & ~1;
    // 10000^(-i2/d) = exp(-i2/d * ln(10000))
    float freq = __expf(-(float)i2 * inv_d * 9.210340372f);
    float ang = (float)pos * freq;
    float pe = (c & 1) ? __cosf(ang) : __sinf(ang);
    float x = emb[(size_t)row * 1024 + c] + pe;
    vals[j] = x;
    s += x;
    s2 += x * x;
  }

  __shared__ float red0[256];
  __shared__ float red1[256];
  red0[t] = s;
  red1[t] = s2;
  __syncthreads();
  for (int off = 128; off > 0; off >>= 1) {
    if (t < off) {
      red0[t] += red0[t + off];
      red1[t] += red1[t + off];
    }
    __syncthreads();
  }
  float mu  = red0[0] * inv_d;
  float var = red1[0] * inv_d - mu * mu;
  float rinv = rsqrtf(var + 1e-5f);

#pragma unroll
  for (int j = 0; j < 4; ++j) {
    int c = t * 4 + j;
    float y = (vals[j] - mu) * rinv * gamma[c] + beta[c];
    xh[(size_t)row * 1024 + c] = (_Float16)y;
  }
}

// ---------------------------------------------------------------------------
// Kernel 2: Wt[j][k] = (f16) W[k][j]   (1024 x 1024)
// ---------------------------------------------------------------------------
__global__ void wconv_kernel(const float* __restrict__ W, _Float16* __restrict__ Wt) {
  int idx = blockIdx.x * 256 + threadIdx.x;   // 0 .. 1M-1
  int k = idx >> 10;
  int j = idx & 1023;
  Wt[(size_t)j * 1024 + k] = (_Float16)W[idx];
}

// ---------------------------------------------------------------------------
// Kernel 3: GEMM  out = xh @ W + bias  (4096 x 1024) via WMMA f16->f32
// wave computes 16(M) x 64(N) tile; 8 waves / block; grid.x = 512
// Software-pipelined: K-step kk+32 fragments are loaded into distinct
// registers while the 4 WMMAs of step kk execute (partial loadcnt waits).
// transposed_v: store result as vt[b][col][token] (for attention PV B-frags)
// ---------------------------------------------------------------------------
__global__ void qkv_gemm_kernel(const _Float16* __restrict__ xh,
                                const _Float16* __restrict__ wt,   // [n][k] f16
                                const float* __restrict__ bias,
                                _Float16* __restrict__ outh,
                                int transposed_v) {
  const int lane = threadIdx.x & 31;
  const int wv = threadIdx.x >> 5;
  const int wid = blockIdx.x * 8 + wv;   // 0..4095
  const int mTile = wid >> 4;            // 0..255
  const int nGroup = wid & 15;           // 0..15
  const int rowBase = mTile * 16;
  const int nBase = nGroup * 64;
  const int hf = lane >> 4;              // 0/1: half-wave
  const int ln = lane & 15;

  const _Float16* arow  = xh + (size_t)(rowBase + ln) * 1024;
  const _Float16* brow0 = wt + (size_t)(nBase + 0 * 16 + ln) * 1024;
  const _Float16* brow1 = wt + (size_t)(nBase + 1 * 16 + ln) * 1024;
  const _Float16* brow2 = wt + (size_t)(nBase + 2 * 16 + ln) * 1024;
  const _Float16* brow3 = wt + (size_t)(nBase + 3 * 16 + ln) * 1024;

  const int lo = hf * 8;
  const int hi = 16 + hf * 8;

  v8f acc0 = zero8f(), acc1 = zero8f(), acc2 = zero8f(), acc3 = zero8f();

  // ---- prologue: load K-step 0 ----
  v16h a  = ld_frag(arow,  lo, hi);
  v16h b0 = ld_frag(brow0, lo, hi);
  v16h b1 = ld_frag(brow1, lo, hi);
  v16h b2 = ld_frag(brow2, lo, hi);
  v16h b3 = ld_frag(brow3, lo, hi);

  // ---- steady state: prefetch kk+32 while computing kk ----
  for (int kk = 0; kk < 1024 - 32; kk += 32) {
    const int nlo = kk + 32 + lo;
    const int nhi = kk + 32 + hi;
    v16h an  = ld_frag(arow,  nlo, nhi);
    v16h b0n = ld_frag(brow0, nlo, nhi);
    v16h b1n = ld_frag(brow1, nlo, nhi);
    v16h b2n = ld_frag(brow2, nlo, nhi);
    v16h b3n = ld_frag(brow3, nlo, nhi);

    acc0 = WMMA_F16(a, b0, acc0);
    acc1 = WMMA_F16(a, b1, acc1);
    acc2 = WMMA_F16(a, b2, acc2);
    acc3 = WMMA_F16(a, b3, acc3);

    a = an; b0 = b0n; b1 = b1n; b2 = b2n; b3 = b3n;
  }
  // ---- epilogue: last K-step ----
  acc0 = WMMA_F16(a, b0, acc0);
  acc1 = WMMA_F16(a, b1, acc1);
  acc2 = WMMA_F16(a, b2, acc2);
  acc3 = WMMA_F16(a, b3, acc3);

  // C/D layout: VGPR r -> row M = r + 8*hf, col N = ln (within 16x16 tile)
  if (!transposed_v) {
#pragma unroll
    for (int nt = 0; nt < 4; ++nt) {
      const v8f& acc = (nt == 0) ? acc0 : (nt == 1) ? acc1 : (nt == 2) ? acc2 : acc3;
      int col = nBase + nt * 16 + ln;
      float bv = bias[col];
#pragma unroll
      for (int r = 0; r < 8; ++r) {
        int row = rowBase + r + hf * 8;
        outh[(size_t)row * 1024 + col] = (_Float16)(acc[r] + bv);
      }
    }
  } else {
    // vt[b][col][token]; tokens r=0..7 are consecutive -> one b128 store
    const int rowStart = rowBase + hf * 8;
    const int bb = rowStart >> 10;
    const int tok = rowStart & 1023;
#pragma unroll
    for (int nt = 0; nt < 4; ++nt) {
      const v8f& acc = (nt == 0) ? acc0 : (nt == 1) ? acc1 : (nt == 2) ? acc2 : acc3;
      int col = nBase + nt * 16 + ln;
      float bv = bias[col];
      v8h pack;
#pragma unroll
      for (int r = 0; r < 8; ++r) pack[r] = (_Float16)(acc[r] + bv);
      *(v8h*)(outh + ((size_t)bb * 1024 + col) * 1024 + tok) = pack;
    }
  }
}

// ---------------------------------------------------------------------------
// Kernel 4: flash attention. wave handles one (b, h, 16-query tile).
// grid.x = 512 blocks x 256 threads (8 waves) -> 4096 waves.
// All 16 global loads (K frags + V frags) issue as one batch at the top of
// each key-block iteration; score WMMAs take a partial loadcnt wait and V
// arrives under the softmax VALU work + LDS round-trip.
// ---------------------------------------------------------------------------
__global__ void attn_kernel(const _Float16* __restrict__ qh,
                            const _Float16* __restrict__ kh,
                            const _Float16* __restrict__ vt,
                            float* __restrict__ out) {
  __shared__ _Float16 pbuf[8][16 * 32];   // per-wave P staging (f32 C-layout -> f16 A-layout)

  const int lane = threadIdx.x & 31;
  const int wv = threadIdx.x >> 5;
  const int wid = blockIdx.x * 8 + wv;    // 0..4095
  const int qTile = wid & 63;
  const int h = (wid >> 6) & 15;
  const int b = wid >> 10;
  const int hf = lane >> 4;
  const int ln = lane & 15;

  // Q tile as two A-fragments (hd 0..31, 32..63)
  const _Float16* qrow = qh + ((size_t)(b * 1024 + qTile * 16 + ln)) * 1024 + h * 64;
  v16h qa0 = ld_frag(qrow, hf * 8, 16 + hf * 8);
  v16h qa1 = ld_frag(qrow, 32 + hf * 8, 48 + hf * 8);

  const _Float16* kbase = kh + (size_t)b * 1024 * 1024 + (size_t)h * 64;
  const _Float16* vrow0 = vt + ((size_t)b * 1024 + (size_t)h * 64 + 0 * 16 + ln) * 1024;
  const _Float16* vrow1 = vrow0 + (size_t)16 * 1024;
  const _Float16* vrow2 = vrow0 + (size_t)32 * 1024;
  const _Float16* vrow3 = vrow0 + (size_t)48 * 1024;
  _Float16* pb = pbuf[wv];

  v8f o0 = zero8f(), o1 = zero8f(), o2 = zero8f(), o3 = zero8f();
  float mrow[8], lrow[8];
#pragma unroll
  for (int r = 0; r < 8; ++r) { mrow[r] = -1e30f; lrow[r] = 0.0f; }

  for (int kb = 0; kb < 1024; kb += 32) {
    // ----- batch all global loads for this key block -----
    const _Float16* krow0 = kbase + (size_t)(kb + ln) * 1024;
    const _Float16* krow1 = kbase + (size_t)(kb + 16 + ln) * 1024;
    v16h b00 = ld_frag(krow0, hf * 8, 16 + hf * 8);        // K, hd 0..31
    v16h b01 = ld_frag(krow0, 32 + hf * 8, 48 + hf * 8);   // K, hd 32..63
    v16h b10 = ld_frag(krow1, hf * 8, 16 + hf * 8);
    v16h b11 = ld_frag(krow1, 32 + hf * 8, 48 + hf * 8);
    v16h vb0 = ld_frag(vrow0, kb + hf * 8, kb + 16 + hf * 8);  // V (indep of scores)
    v16h vb1 = ld_frag(vrow1, kb + hf * 8, kb + 16 + hf * 8);
    v16h vb2 = ld_frag(vrow2, kb + hf * 8, kb + 16 + hf * 8);
    v16h vb3 = ld_frag(vrow3, kb + hf * 8, kb + 16 + hf * 8);

    // ----- scores for 32 keys: two 16-key tiles -----
    v8f s0 = zero8f(), s1 = zero8f();
    s0 = WMMA_F16(qa0, b00, s0);
    s0 = WMMA_F16(qa1, b01, s0);
    s1 = WMMA_F16(qa0, b10, s1);
    s1 = WMMA_F16(qa1, b11, s1);

    // ----- online softmax (row = r + 8*hf, reductions across 16-lane groups) -----
    float corr[8];
#pragma unroll
    for (int r = 0; r < 8; ++r) {
      float a0 = s0[r] * 0.125f;       // 1/sqrt(64)
      float a1 = s1[r] * 0.125f;
      float mx = fmaxf(a0, a1);
#pragma unroll
      for (int msk = 8; msk > 0; msk >>= 1) mx = fmaxf(mx, __shfl_xor(mx, msk, 16));
      float mnew = fmaxf(mrow[r], mx);
      float c = __expf(mrow[r] - mnew);
      float p0 = __expf(a0 - mnew);
      float p1 = __expf(a1 - mnew);
      float ps = p0 + p1;
#pragma unroll
      for (int msk = 8; msk > 0; msk >>= 1) ps += __shfl_xor(ps, msk, 16);
      lrow[r] = lrow[r] * c + ps;
      mrow[r] = mnew;
      corr[r] = c;
      s0[r] = p0;
      s1[r] = p1;
    }
#pragma unroll
    for (int r = 0; r < 8; ++r) {
      o0[r] *= corr[r]; o1[r] *= corr[r]; o2[r] *= corr[r]; o3[r] *= corr[r];
    }

    // ----- P (f32 C-layout) -> LDS -> f16 A-layout fragment -----
    asm volatile("s_wait_dscnt 0x0" ::: "memory");   // prior-iter LDS reads retired
    __builtin_amdgcn_wave_barrier();
#pragma unroll
    for (int r = 0; r < 8; ++r) {
      int m = r + hf * 8;
      pb[m * 32 + ln]      = (_Float16)s0[r];
      pb[m * 32 + 16 + ln] = (_Float16)s1[r];
    }
    asm volatile("s_wait_dscnt 0x0" ::: "memory");   // stores visible wave-wide
    __builtin_amdgcn_wave_barrier();
    v16h pa = ld_frag(pb + ln * 32, hf * 8, 16 + hf * 8);

    // ----- O += P @ V -----
    o0 = WMMA_F16(pa, vb0, o0);
    o1 = WMMA_F16(pa, vb1, o1);
    o2 = WMMA_F16(pa, vb2, o2);
    o3 = WMMA_F16(pa, vb3, o3);
  }

  // ----- epilogue: divide by softmax denominator, store f32 -----
  size_t obase = ((size_t)(b * 1024 + qTile * 16)) * 1024 + (size_t)h * 64;
#pragma unroll
  for (int r = 0; r < 8; ++r) {
    float inv = 1.0f / lrow[r];
    size_t rowoff = obase + (size_t)(r + hf * 8) * 1024 + ln;
    out[rowoff]      = o0[r] * inv;
    out[rowoff + 16] = o1[r] * inv;
    out[rowoff + 32] = o2[r] * inv;
    out[rowoff + 48] = o3[r] * inv;
  }
}

// ---------------------------------------------------------------------------
extern "C" void kernel_launch(void* const* d_in, const int* in_sizes, int n_in,
                              void* d_out, int out_size, void* d_ws, size_t ws_size,
                              hipStream_t stream) {
  const float* emb   = (const float*)d_in[0];
  const float* gamma = (const float*)d_in[1];
  const float* beta  = (const float*)d_in[2];
  const float* Wq    = (const float*)d_in[3];
  const float* bq    = (const float*)d_in[4];
  const float* Wk    = (const float*)d_in[5];
  const float* bk    = (const float*)d_in[6];
  const float* Wv    = (const float*)d_in[7];
  const float* bv    = (const float*)d_in[8];
  float* out = (float*)d_out;

  char* ws = (char*)d_ws;
  _Float16* xh  = (_Float16*)(ws);                              // 8 MB
  _Float16* wtq = (_Float16*)(ws + (size_t)8 * 1024 * 1024);    // 2 MB
  _Float16* wtk = wtq + (size_t)1024 * 1024;                    // 2 MB
  _Float16* wtv = wtk + (size_t)1024 * 1024;                    // 2 MB
  _Float16* qh  = (_Float16*)(ws + (size_t)14 * 1024 * 1024);   // 8 MB
  _Float16* kh2 = (_Float16*)(ws + (size_t)22 * 1024 * 1024);   // 8 MB
  _Float16* vt  = (_Float16*)(ws + (size_t)30 * 1024 * 1024);   // 8 MB

  pe_ln_kernel<<<4096, 256, 0, stream>>>(emb, gamma, beta, xh);

  wconv_kernel<<<4096, 256, 0, stream>>>(Wq, wtq);
  wconv_kernel<<<4096, 256, 0, stream>>>(Wk, wtk);
  wconv_kernel<<<4096, 256, 0, stream>>>(Wv, wtv);

  qkv_gemm_kernel<<<512, 256, 0, stream>>>(xh, wtq, bq, qh, 0);
  qkv_gemm_kernel<<<512, 256, 0, stream>>>(xh, wtk, bk, kh2, 0);
  qkv_gemm_kernel<<<512, 256, 0, stream>>>(xh, wtv, bv, vt, 1);

  attn_kernel<<<512, 256, 0, stream>>>(qh, kh2, vt, out);
}
